// STFACH_7138235646200
// MI455X (gfx1250) — matrix operand order repaired and checked by reading, not compiled, not run on hardware
//
#include <hip/hip_runtime.h>

typedef __attribute__((ext_vector_type(16))) __bf16 v16bf;
typedef __attribute__((ext_vector_type(8)))  float  v8f;
typedef __attribute__((ext_vector_type(8)))  unsigned int v8u;
typedef __attribute__((ext_vector_type(4)))  unsigned int u32x4;
typedef __attribute__((ext_vector_type(8)))  int i32x8;
typedef __attribute__((ext_vector_type(4)))  int i32x4;

#define DEVINL __device__ __forceinline__

#if __has_builtin(__builtin_amdgcn_tensor_load_to_lds)
#define HAVE_TDM 1
#endif

// ---------------------------------------------------------------- helpers ---
DEVINL unsigned short f2bf(float f) {
  unsigned int u = __float_as_uint(f);
  u += 0x7FFFu + ((u >> 16) & 1u);   // round-to-nearest-even
  return (unsigned short)(u >> 16);
}

// A fragment 16x32 bf16 from row-major [rows][ldk] (K contiguous).
// Contiguous k order (B uses the same): lane m = lane&15; VGPR p holds K = 16*(lane>=16) + 2p (+1).
DEVINL v16bf load_fragA(const unsigned short* base, int row0, int ldk, int k0, int kvalid) {
  int lane = threadIdx.x & 31;
  int hi = (lane >> 4) & 1;
  const unsigned short* p0 = base + (size_t)(row0 + (lane & 15)) * ldk + k0;
  v8u r;
#pragma unroll
  for (int p = 0; p < 8; ++p) {
    int kb = (hi << 4) + (p << 1);
    r[p] = (kb < kvalid) ? *(const unsigned int*)(p0 + kb) : 0u;
  }
  return __builtin_bit_cast(v16bf, r);
}

// B fragment 32x16 bf16 from [N][K] storage (pre-transposed weight, K contiguous).
// Contiguous k order, same as A (B row striped across lanes): col n = lane&15; VGPR p holds K = 16*(lane>=16) + 2p (+1).
DEVINL v16bf load_fragB_nk(const unsigned short* base, int col0, int ldk, int k0, int kvalid) {
  int lane = threadIdx.x & 31;
  int hi = (lane >> 4) & 1;
  const unsigned short* p0 = base + (size_t)(col0 + (lane & 15)) * ldk + k0 + (hi << 4);
  v8u r;
#pragma unroll
  for (int p = 0; p < 8; ++p) {
    int kb = (hi << 4) + (p << 1);
    r[p] = (kb < kvalid) ? *(const unsigned int*)(p0 + (p << 1)) : 0u;
  }
  return __builtin_bit_cast(v16bf, r);
}

// B fragment 32x16 bf16 from [K][N] storage (K-major activation matrix).
DEVINL v16bf load_fragB_kn(const unsigned short* base, int col0, int ldn, int k0, int kvalid) {
  int lane = threadIdx.x & 31;
  int hi = (lane >> 4) & 1;
  int col = col0 + (lane & 15);
  v8u r;
#pragma unroll
  for (int p = 0; p < 8; ++p) {
    int kb = (hi << 4) + (p << 1);
    unsigned int lo = 0u, h2 = 0u;
    if (kb < kvalid) {
      lo = base[(size_t)(k0 + kb) * ldn + col];
      h2 = base[(size_t)(k0 + kb + 1) * ldn + col];
    }
    r[p] = lo | (h2 << 16);
  }
  return __builtin_bit_cast(v16bf, r);
}

DEVINL v8f wmma_bf16(v16bf a, v16bf b, v8f c) {
  return __builtin_amdgcn_wmma_f32_16x16x32_bf16(false, a, false, b, (short)0, c, false, false);
}

// TDM: async 1D copy of `qwords` 8-byte elements from global to LDS.
// D# per CDNA5 ISA: group0 {count=1, lds_addr, global_addr, type=2},
// group1 {mask=0, data_size=3 (8B), dim0=tile0=stride0=qwords, dim1=tile1=1}, groups 2/3 zero.
DEVINL void tdm_load_1d(const void* gsrc, unsigned lds_byte_addr, unsigned qwords) {
#ifdef HAVE_TDM
  unsigned long long ga = (unsigned long long)gsrc;
  u32x4 g0;
  g0[0] = 1u;                                                  // count=1 (valid user D#)
  g0[1] = lds_byte_addr;                                       // LDS byte address
  g0[2] = (unsigned)ga;                                        // global addr [31:0]
  g0[3] = ((unsigned)(ga >> 32) & 0x01FFFFFFu) | (2u << 30);   // global addr [56:32] | type=2
  i32x8 g1;
  g1[0] = (int)(3u << 16);                                     // data_size = 8B
  g1[1] = (int)((qwords & 0xFFFFu) << 16);                     // tensor_dim0 [15:0]
  g1[2] = (int)((qwords >> 16) | (1u << 16));                  // tensor_dim0 [31:16] | tensor_dim1 lo = 1
  g1[3] = (int)((qwords & 0xFFFFu) << 16);                     // tile_dim0
  g1[4] = 1;                                                   // tile_dim1 = 1
  g1[5] = (int)qwords;                                         // tensor_dim0_stride [31:0]
  g1[6] = 0;
  g1[7] = 0;
  i32x4 g2 = {0, 0, 0, 0};
  i32x4 g3 = {0, 0, 0, 0};
#if defined(__clang_major__) && (__clang_major__ >= 23)
  i32x8 g4 = {0, 0, 0, 0, 0, 0, 0, 0};
  __builtin_amdgcn_tensor_load_to_lds(g0, g1, g2, g3, g4, 0);
#else
  __builtin_amdgcn_tensor_load_to_lds(g0, g1, g2, g3, 0);
#endif
#else
  (void)gsrc; (void)lds_byte_addr; (void)qwords;
#endif
}

// ------------------------------------------------------------ pack kernels ---
// dst[n*Kpad+k] = src[k*N+n] (f32 -> bf16 transpose, zero-padded)
__global__ void k_pack_wT(const float* __restrict__ src, unsigned short* __restrict__ dst,
                          int K, int N, int Kpad, int Npad) {
  int i = blockIdx.x * 256 + threadIdx.x;
  if (i >= Kpad * Npad) return;
  int k = i % Kpad, n = i / Kpad;
  float v = (k < K && n < N) ? src[(size_t)k * N + n] : 0.f;
  dst[i] = f2bf(v);
}

// adj f32 [307][307] -> bf16 [320][320] zero padded (used as A matrix, no transpose)
__global__ void k_pack_adj(const float* __restrict__ a, unsigned short* __restrict__ dst) {
  int i = blockIdx.x * 256 + threadIdx.x;
  if (i >= 320 * 320) return;
  int m = i % 320, n = i / 320;
  dst[i] = f2bf((n < 307 && m < 307) ? a[(size_t)n * 307 + m] : 0.f);
}

// Wg_cat^T bf16 [64][320]: k<152 -> Wg1[k][n]; 160<=k<312 -> Wg2[k-160][n]; else 0
__global__ void k_pack_wgcat(const float* __restrict__ Wg1, const float* __restrict__ Wg2,
                             unsigned short* __restrict__ dst) {
  int i = blockIdx.x * 256 + threadIdx.x;
  if (i >= 64 * 320) return;
  int k = i % 320, n = i / 320;
  float v = 0.f;
  if (k < 152) v = Wg1[(size_t)k * 64 + n];
  else if (k >= 160 && k < 312) v = Wg2[(size_t)(k - 160) * 64 + n];
  dst[i] = f2bf(v);
}

// ---------------------------------------------------------- embedding etc ---
// writes e1 bf16 [B*N pad1232][1824] (rows b*307+n, cols t*152+f)
// and    e2 bf16 [B*T][320][160]    (rows node m, cols feature f)
__global__ void k_embed(const float* __restrict__ ori, const float* __restrict__ Wv,
                        const float* __restrict__ bv, const float* __restrict__ tode,
                        const float* __restrict__ dowe, const float* __restrict__ nodee,
                        unsigned short* __restrict__ e1, unsigned short* __restrict__ e2) {
  int i = blockIdx.x * 256 + threadIdx.x;
  if (i >= 4 * 12 * 307 * 152) return;
  int f = i % 152;
  int r = i / 152;
  int n = r % 307;
  int r2 = r / 307;
  int t = r2 % 12;
  int b = r2 / 12;
  const float* x = ori + ((size_t)(b * 12 + t) * 307 + n) * 3;
  float val;
  if (f < 24) {
    val = bv[f] + x[0] * Wv[f] + x[1] * Wv[24 + f] + x[2] * Wv[48 + f];
  } else if (f < 48) {
    float fr = x[1] - floorf(x[1]);
    int tod = (int)(fr * 288.f);
    tod = tod < 0 ? 0 : (tod > 287 ? 287 : tod);
    val = tode[tod * 24 + (f - 24)];
  } else if (f < 72) {
    float fr = x[2] - floorf(x[2]);
    int dw = (int)(fr * 7.f);
    dw = dw < 0 ? 0 : (dw > 6 ? 6 : dw);
    val = dowe[dw * 24 + (f - 48)];
  } else {
    val = nodee[n * 80 + (f - 72)];
  }
  unsigned short bfv = f2bf(val);
  e1[((size_t)(b * 307 + n)) * 1824 + t * 152 + f] = bfv;
  e2[(((size_t)(b * 12 + t)) * 320 + n) * 160 + f] = bfv;
}

// ce[b][n][t][h] f32 = coeff2 @ W_ce + b_ce
__global__ void k_ce(const float* __restrict__ c2, const float* __restrict__ Wce,
                     const float* __restrict__ bce, float* __restrict__ ce) {
  int i = blockIdx.x * 256 + threadIdx.x;
  if (i >= 4 * 307 * 11 * 64) return;
  int h = i & 63;
  const float* c = c2 + (size_t)(i >> 6) * 2;
  ce[i] = c[0] * Wce[h] + c[1] * Wce[64 + h] + bce[h];
}

// stage z [B,N,S,H] f32 -> zbf [B,S,320,64] bf16 (pad rows stay zero)
__global__ void k_zbf(const float* __restrict__ z, unsigned short* __restrict__ zbf) {
  int i = blockIdx.x * 256 + threadIdx.x;
  if (i >= 4 * 11 * 307 * 64) return;
  int h = i & 63;
  int r = i >> 6;
  int n = r % 307;
  int r2 = r / 307;
  int s = r2 % 11;
  int b = r2 / 11;
  zbf[(((size_t)(b * 11 + s)) * 320 + n) * 64 + h] =
      f2bf(z[(((size_t)(b * 307 + n)) * 11 + s) * 64 + h]);
}

// -------------------------------------------------------------- big GEMMs ---
// x_in: [1232x1824] @ W_in^T-> z [1228][704]  (z init = x_in)
__global__ void k_xin(const unsigned short* __restrict__ e1, const unsigned short* __restrict__ WinT,
                      const float* __restrict__ b_in, float* __restrict__ z) {
  int task = blockIdx.x * 8 + (threadIdx.x >> 5);
  if (task >= 77 * 44) return;
  int mt = task % 77, ct = task / 77;
  v8f acc = {};
  for (int ks = 0; ks < 57; ++ks) {
    v16bf a = load_fragA(e1, mt * 16, 1824, ks * 32, 32);
    v16bf w = load_fragB_nk(WinT, ct * 16, 1824, ks * 32, 32);
    acc = wmma_bf16(a, w, acc);
  }
  int lane = threadIdx.x & 31, hi = lane >> 4, cl = lane & 15;
  int col = ct * 16 + cl;
  float bias = b_in[col];
#pragma unroll
  for (int r = 0; r < 8; ++r) {
    int row = mt * 16 + r + hi * 8;
    if (row < 1228) z[(size_t)row * 704 + col] = acc[r] + bias;
  }
}

// qkv: zbf[b,s] [320x64] @ {Wq,Wk,Wv}^T -> q/k/v bf16 [b,s][320][64]
__global__ void k_qkv(const unsigned short* __restrict__ zbf,
                      const unsigned short* __restrict__ WqT, const unsigned short* __restrict__ WkT,
                      const unsigned short* __restrict__ WvT,
                      const float* __restrict__ bq, const float* __restrict__ bk,
                      const float* __restrict__ bv,
                      unsigned short* __restrict__ q, unsigned short* __restrict__ k,
                      unsigned short* __restrict__ v) {
  int task = blockIdx.x * 8 + (threadIdx.x >> 5);
  if (task >= 4 * 11 * 20 * 4 * 3) return;
  int mat = task % 3;
  int ct = (task / 3) % 4;
  int mt = (task / 12) % 20;
  int bs = task / 240;
  const unsigned short* WT = mat == 0 ? WqT : (mat == 1 ? WkT : WvT);
  const float* bias = mat == 0 ? bq : (mat == 1 ? bk : bv);
  unsigned short* out = mat == 0 ? q : (mat == 1 ? k : v);
  const unsigned short* A = zbf + (size_t)bs * 320 * 64;
  v8f acc = {};
#pragma unroll
  for (int ks = 0; ks < 2; ++ks) {
    v16bf a = load_fragA(A, mt * 16, 64, ks * 32, 32);
    v16bf w = load_fragB_nk(WT, ct * 16, 64, ks * 32, 32);
    acc = wmma_bf16(a, w, acc);
  }
  int lane = threadIdx.x & 31, hi = lane >> 4, cl = lane & 15;
  int col = ct * 16 + cl;
  float bb = bias[col];
#pragma unroll
  for (int r = 0; r < 8; ++r) {
    int row = mt * 16 + r + hi * 8;
    out[((size_t)bs * 320 + row) * 64 + col] = f2bf(acc[r] + bb);
  }
}

// GCN aggregation: agg_cat[bt][320][320] (cols 0..159 adj, 160..319 adj_sem)
__global__ void k_agg(const unsigned short* __restrict__ adjA, const unsigned short* __restrict__ adjS,
                      const unsigned short* __restrict__ e2, unsigned short* __restrict__ agg) {
  int task = blockIdx.x * 8 + (threadIdx.x >> 5);
  if (task >= 48 * 20 * 10 * 2) return;
  int asel = task & 1;
  int ct = (task >> 1) % 10;
  int mt = (task / 20) % 20;
  int bt = task / 400;
  const unsigned short* A = asel ? adjS : adjA;
  const unsigned short* B = e2 + (size_t)bt * 320 * 160;
  v8f acc = {};
  for (int ks = 0; ks < 10; ++ks) {
    v16bf a = load_fragA(A, mt * 16, 320, ks * 32, 32);
    v16bf b = load_fragB_kn(B, ct * 16, 160, ks * 32, 32);
    acc = wmma_bf16(a, b, acc);
  }
  int lane = threadIdx.x & 31, hi = lane >> 4, cl = lane & 15;
  int col = asel * 160 + ct * 16 + cl;
#pragma unroll
  for (int r = 0; r < 8; ++r)
    agg[((size_t)bt * 320 + mt * 16 + r + hi * 8) * 320 + col] = f2bf(acc[r]);
}

// GCN proj: g = relu(agg_cat @ Wg_cat + bg) -> g_p bf16 [b*307+n][t*64+h]
__global__ void k_gcnproj(const unsigned short* __restrict__ agg, const unsigned short* __restrict__ WgT,
                          const float* __restrict__ bg, unsigned short* __restrict__ g_p) {
  int task = blockIdx.x * 8 + (threadIdx.x >> 5);
  if (task >= 960 * 4) return;
  int mt = task % 960, ct = task / 960;
  v8f acc = {};
  for (int ks = 0; ks < 10; ++ks) {
    v16bf a = load_fragA(agg, mt * 16, 320, ks * 32, 32);
    v16bf w = load_fragB_nk(WgT, ct * 16, 320, ks * 32, 32);
    acc = wmma_bf16(a, w, acc);
  }
  int lane = threadIdx.x & 31, hi = lane >> 4, cl = lane & 15;
  int col = ct * 16 + cl;
  float bb = bg[col];
#pragma unroll
  for (int r = 0; r < 8; ++r) {
    int grow = mt * 16 + r + hi * 8;
    int bt = grow / 320, m = grow % 320;
    if (m < 307) {
      int b = bt / 12, t = bt % 12;
      float val = fmaxf(acc[r] + bb, 0.f);
      g_p[((size_t)(b * 307 + m)) * 768 + t * 64 + col] = f2bf(val);
    }
  }
}

// gp: g_p [1232x768] @ Wgo^T[16x768] -> out_het f32 [1232][16]
__global__ void k_gp(const unsigned short* __restrict__ g_p, const unsigned short* __restrict__ WgoT,
                     const float* __restrict__ bgo, float* __restrict__ out_het) {
  int task = blockIdx.x * 8 + (threadIdx.x >> 5);
  if (task >= 77) return;
  int mt = task;
  v8f acc = {};
  for (int ks = 0; ks < 24; ++ks) {
    v16bf a = load_fragA(g_p, mt * 16, 768, ks * 32, 32);
    v16bf w = load_fragB_nk(WgoT, 0, 768, ks * 32, 32);
    acc = wmma_bf16(a, w, acc);
  }
  int lane = threadIdx.x & 31, hi = lane >> 4, cl = lane & 15;
  float bb = (cl < 12) ? bgo[cl] : 0.f;
#pragma unroll
  for (int r = 0; r < 8; ++r)
    out_het[(size_t)(mt * 16 + r + hi * 8) * 16 + cl] = acc[r] + bb;
}

// ---------------- fused attention + FFN + LN + Euler update (one step) ------
// Dynamic LDS map (124928 B total):
//   [0, 81920)        : scores f32 [64 rows][320]; P bf16 written in-place per row (ld=640)
//   [49152, 67584)    : h1bf/h2bf/hbuf/h2buf (reused after P is consumed)
//   [81920, 83968)    : obf [16][64] bf16
//   [83968, 124928)   : V tile [320][64] bf16, filled asynchronously by TDM
__global__ void k_attn(const unsigned short* __restrict__ qm, const unsigned short* __restrict__ km,
                       const unsigned short* __restrict__ vm, float* __restrict__ z,
                       const float* __restrict__ ce,
                       const unsigned short* __restrict__ WoT, const unsigned short* __restrict__ Wf1T,
                       const unsigned short* __restrict__ Wf2T,
                       const float* __restrict__ bo, const float* __restrict__ ln1g,
                       const float* __restrict__ ln1b, const float* __restrict__ bf1,
                       const float* __restrict__ bf2, const float* __restrict__ ln2g,
                       const float* __restrict__ ln2b, const float* __restrict__ Wtp2,
                       const float* __restrict__ btp2, int stepi) {
  extern __shared__ char smem[];
  float* sc = (float*)smem;
  unsigned short* pb = (unsigned short*)smem;
  unsigned short* h1bf = (unsigned short*)(smem + 49152);
  unsigned short* h2bf = (unsigned short*)(smem + 49152 + 2048);
  float* hbuf = (float*)(smem + 49152 + 10240);
  float* h2buf = (float*)(smem + 49152 + 14336);
  unsigned short* obf = (unsigned short*)(smem + 81920);
  unsigned short* vl = (unsigned short*)(smem + 83968);

  int blk = blockIdx.x;
  int qt = blk % 20;
  int s = (blk / 20) % 11;
  int b = blk / 220;
  int bs = b * 11 + s;
  const unsigned short* qb = qm + (size_t)bs * 320 * 64;
  const unsigned short* kb = km + (size_t)bs * 320 * 64;
  const unsigned short* vb = vm + (size_t)bs * 320 * 64;
  int lane = threadIdx.x & 31, wave = threadIdx.x >> 5, hi = lane >> 4, cl = lane & 15;

  // ---- async: start TDM copy of V tile (40 KB) into LDS; overlaps stages 1-2
#ifdef HAVE_TDM
  if (wave == 0)
    tdm_load_1d(vb, (unsigned)(unsigned long long)(size_t)vl, 40960u / 8u);
#else
  for (int i = threadIdx.x; i < 5120; i += 256)
    ((unsigned long long*)vl)[i] = ((const unsigned long long*)vb)[i];
#endif

  // ---- stage 1: scores S = Q K^T / 4  (80 tile-tasks)
  for (int it = 0; it < 10; ++it) {
    int task = wave + it * 8;
    int h = task / 20, kt = task % 20;
    v16bf a = load_fragA(qb, qt * 16, 64, h * 16, 16);
    v16bf bf = load_fragB_nk(kb, kt * 16, 64, h * 16, 16);
    v8f c = {};
    c = wmma_bf16(a, bf, c);
#pragma unroll
    for (int r = 0; r < 8; ++r)
      sc[(h * 16 + r + hi * 8) * 320 + kt * 16 + cl] = c[r] * 0.25f;
  }
  __syncthreads();

  // ---- stage 2: row softmax over keys (mask m>=307), write P bf16 in place
  if (threadIdx.x < 64) {
    float* srow = sc + threadIdx.x * 320;
    unsigned short* prow = pb + threadIdx.x * 640;
    float mx = -1e30f;
    for (int m = 0; m < 307; ++m) mx = fmaxf(mx, srow[m]);
    float sum = 0.f;
    for (int m = 0; m < 307; ++m) {
      float e = __expf(srow[m] - mx);
      srow[m] = e;
      sum += e;
    }
    float inv = 1.f / sum;
    for (int m = 0; m < 307; ++m) prow[m] = f2bf(srow[m] * inv);
    for (int m = 307; m < 320; ++m) prow[m] = 0;
  }
#ifdef HAVE_TDM
  if (wave == 0) __builtin_amdgcn_s_wait_tensorcnt(0);
#endif
  __syncthreads();

  // ---- stage 3: O = P @ V per head (V from LDS)
  if (wave < 4) {
    int h = wave;
    v8f acc = {};
    for (int ks = 0; ks < 10; ++ks) {
      v16bf a = load_fragA(pb, h * 16, 640, ks * 32, 32);
      v16bf bf = load_fragB_kn(vl, h * 16, 64, ks * 32, 32);
      acc = wmma_bf16(a, bf, acc);
    }
#pragma unroll
    for (int r = 0; r < 8; ++r)
      obf[(r + hi * 8) * 64 + h * 16 + cl] = f2bf(acc[r]);
  }
  __syncthreads();

  // ---- stage 4: o @ Wo + bo + residual z
  if (wave < 4) {
    int ct = wave;
    v8f acc = {};
#pragma unroll
    for (int ks = 0; ks < 2; ++ks) {
      v16bf a = load_fragA(obf, 0, 64, ks * 32, 32);
      v16bf bf = load_fragB_nk(WoT, ct * 16, 64, ks * 32, 32);
      acc = wmma_bf16(a, bf, acc);
    }
    int col = ct * 16 + cl;
    float bb = bo[col];
#pragma unroll
    for (int r = 0; r < 8; ++r) {
      int row = r + hi * 8;
      int n = qt * 16 + row;
      float zt = (n < 307) ? z[(((size_t)(b * 307 + n)) * 11 + s) * 64 + col] : 0.f;
      hbuf[row * 64 + col] = acc[r] + bb + zt;
    }
  }
  __syncthreads();

  // ---- stage 5: LN1
  if (threadIdx.x < 16) {
    int row = threadIdx.x;
    float* x = hbuf + row * 64;
    float m = 0.f;
    for (int h = 0; h < 64; ++h) m += x[h];
    m *= (1.f / 64.f);
    float v2 = 0.f;
    for (int h = 0; h < 64; ++h) { float d = x[h] - m; v2 += d * d; }
    v2 *= (1.f / 64.f);
    float rs = rsqrtf(v2 + 1e-5f);
    for (int h = 0; h < 64; ++h) {
      float val = (x[h] - m) * rs * ln1g[h] + ln1b[h];
      x[h] = val;
      h1bf[row * 64 + h] = f2bf(val);
    }
  }
  __syncthreads();

  // ---- stage 6: FFN1 relu(h1 @ Wf1 + bf1) -> [16][256] bf16
  for (int it = 0; it < 2; ++it) {
    int ct = wave + it * 8;
    v8f acc = {};
#pragma unroll
    for (int ks = 0; ks < 2; ++ks) {
      v16bf a = load_fragA(h1bf, 0, 64, ks * 32, 32);
      v16bf bf = load_fragB_nk(Wf1T, ct * 16, 64, ks * 32, 32);
      acc = wmma_bf16(a, bf, acc);
    }
    int col = ct * 16 + cl;
    float bb = bf1[col];
#pragma unroll
    for (int r = 0; r < 8; ++r)
      h2bf[(r + hi * 8) * 256 + col] = f2bf(fmaxf(acc[r] + bb, 0.f));
  }
  __syncthreads();

  // ---- stage 7: FFN2 + residual h1
  if (wave < 4) {
    int ct = wave;
    v8f acc = {};
    for (int ks = 0; ks < 8; ++ks) {
      v16bf a = load_fragA(h2bf, 0, 256, ks * 32, 32);
      v16bf bf = load_fragB_nk(Wf2T, ct * 16, 256, ks * 32, 32);
      acc = wmma_bf16(a, bf, acc);
    }
    int col = ct * 16 + cl;
    float bb = bf2[col];
#pragma unroll
    for (int r = 0; r < 8; ++r) {
      int row = r + hi * 8;
      h2buf[row * 64 + col] = acc[r] + bb + hbuf[row * 64 + col];
    }
  }
  __syncthreads();

  // ---- stage 8: LN2 + tanh -> hbuf
  if (threadIdx.x < 16) {
    int row = threadIdx.x;
    float* x = h2buf + row * 64;
    float m = 0.f;
    for (int h = 0; h < 64; ++h) m += x[h];
    m *= (1.f / 64.f);
    float v2 = 0.f;
    for (int h = 0; h < 64; ++h) { float d = x[h] - m; v2 += d * d; }
    v2 *= (1.f / 64.f);
    float rs = rsqrtf(v2 + 1e-5f);
    for (int h = 0; h < 64; ++h)
      hbuf[row * 64 + h] = tanhf((x[h] - m) * rs * ln2g[h] + ln2b[h]);
  }
  __syncthreads();

  // ---- stage 9: Euler update z += f(z) * x2s[stepi]
  for (int idx = threadIdx.x; idx < 1024; idx += 256) {
    int row = idx >> 6, h = idx & 63;
    int n = qt * 16 + row;
    if (n < 307) {
      int c0 = s * 11 + stepi;
      float x2v = btp2[c0];
      const float* cr = ce + ((size_t)(b * 307 + n)) * 11 * 64 + h;
#pragma unroll
      for (int tt = 0; tt < 11; ++tt) x2v += cr[tt * 64] * Wtp2[tt * 121 + c0];
      z[(((size_t)(b * 307 + n)) * 11 + s) * 64 + h] += hbuf[row * 64 + h] * x2v;
    }
  }
}

// ------------------------------------------------- final LN + combine -------
__global__ void k_final(const float* __restrict__ z, const float* __restrict__ out_het,
                        const float* __restrict__ lnfg, const float* __restrict__ lnfb,
                        const float* __restrict__ Wend, const float* __restrict__ bend,
                        const float* __restrict__ wacl, const float* __restrict__ whet,
                        float* __restrict__ out) {
  int i = blockIdx.x * 256 + threadIdx.x;
  if (i >= 4 * 307) return;
  int b = i / 307, n = i % 307;
  const float* zr = z + ((size_t)i * 11 + 10) * 64;
  float m = 0.f;
  for (int h = 0; h < 64; ++h) m += zr[h];
  m *= (1.f / 64.f);
  float v2 = 0.f;
  for (int h = 0; h < 64; ++h) { float d = zr[h] - m; v2 += d * d; }
  v2 *= (1.f / 64.f);
  float rs = rsqrtf(v2 + 1e-5f);
  float zt[64];
  for (int h = 0; h < 64; ++h) zt[h] = (zr[h] - m) * rs * lnfg[h] + lnfb[h];
  float wa = wacl[0], wh = whet[0];
  for (int o = 0; o < 12; ++o) {
    float acc = bend[o];
    for (int h = 0; h < 64; ++h) acc += zt[h] * Wend[h * 12 + o];
    float het = out_het[(size_t)i * 16 + o];
    out[((size_t)b * 12 + o) * 307 + n] = wa * acc + wh * het;
  }
}

// ------------------------------------------------------------------ host ----
extern "C" void kernel_launch(void* const* d_in, const int* in_sizes, int n_in,
                              void* d_out, int out_size, void* d_ws, size_t ws_size,
                              hipStream_t stream) {
  (void)in_sizes; (void)n_in; (void)out_size; (void)ws_size;
  const float* ori    = (const float*)d_in[0];
  const float* coeff2 = (const float*)d_in[2];
  const float* W_val  = (const float*)d_in[5];
  const float* b_val  = (const float*)d_in[6];
  const float* tod_e  = (const float*)d_in[7];
  const float* dow_e  = (const float*)d_in[8];
  const float* node_e = (const float*)d_in[9];
  const float* W_in   = (const float*)d_in[10];
  const float* b_in   = (const float*)d_in[11];
  const float* W_ce   = (const float*)d_in[12];
  const float* b_ce   = (const float*)d_in[13];
  const float* W_tp2  = (const float*)d_in[14];
  const float* b_tp2  = (const float*)d_in[15];
  const float* Wq = (const float*)d_in[20];
  const float* Wk = (const float*)d_in[21];
  const float* Wv = (const float*)d_in[22];
  const float* Wo = (const float*)d_in[23];
  const float* bq = (const float*)d_in[24];
  const float* bk = (const float*)d_in[25];
  const float* bv = (const float*)d_in[26];
  const float* bo = (const float*)d_in[27];
  const float* ln1g = (const float*)d_in[28];
  const float* ln1b = (const float*)d_in[29];
  const float* Wf1  = (const float*)d_in[30];
  const float* bf1  = (const float*)d_in[31];
  const float* Wf2  = (const float*)d_in[32];
  const float* bf2  = (const float*)d_in[33];
  const float* ln2g = (const float*)d_in[34];
  const float* ln2b = (const float*)d_in[35];
  const float* lnfg = (const float*)d_in[36];
  const float* lnfb = (const float*)d_in[37];
  const float* Wend = (const float*)d_in[38];
  const float* bend = (const float*)d_in[39];
  const float* wacl = (const float*)d_in[40];
  const float* whet = (const float*)d_in[41];
  const float* adj  = (const float*)d_in[42];
  const float* adjs = (const float*)d_in[43];
  const float* Wg1  = (const float*)d_in[44];
  const float* Wg2  = (const float*)d_in[45];
  const float* bg   = (const float*)d_in[46];
  const float* Wgo  = (const float*)d_in[47];
  const float* bgo  = (const float*)d_in[48];

  char* ws = (char*)d_ws;
  size_t off = 0;
  auto alloc = [&](size_t bytes) -> char* {
    char* p = ws + off;
    off = (off + bytes + 255) & ~(size_t)255;
    return p;
  };
  unsigned short* e1      = (unsigned short*)alloc(1232ull * 1824 * 2);
  unsigned short* e2      = (unsigned short*)alloc(48ull * 320 * 160 * 2);
  unsigned short* WinT    = (unsigned short*)alloc(704ull * 1824 * 2);
  unsigned short* WqT     = (unsigned short*)alloc(64ull * 64 * 2);
  unsigned short* WkT     = (unsigned short*)alloc(64ull * 64 * 2);
  unsigned short* WvT     = (unsigned short*)alloc(64ull * 64 * 2);
  unsigned short* WoT     = (unsigned short*)alloc(64ull * 64 * 2);
  unsigned short* Wf1T    = (unsigned short*)alloc(256ull * 64 * 2);
  unsigned short* Wf2T    = (unsigned short*)alloc(64ull * 256 * 2);
  unsigned short* WgcatT  = (unsigned short*)alloc(64ull * 320 * 2);
  unsigned short* WgoT    = (unsigned short*)alloc(16ull * 768 * 2);
  unsigned short* adjA    = (unsigned short*)alloc(320ull * 320 * 2);
  unsigned short* adjS    = (unsigned short*)alloc(320ull * 320 * 2);
  float*          zst     = (float*)alloc(1228ull * 704 * 4);
  float*          ce      = (float*)alloc(1228ull * 704 * 4);
  unsigned short* zbf     = (unsigned short*)alloc(44ull * 320 * 64 * 2);
  unsigned short* qb      = (unsigned short*)alloc(44ull * 320 * 64 * 2);
  unsigned short* kbuf    = (unsigned short*)alloc(44ull * 320 * 64 * 2);
  unsigned short* vbuf    = (unsigned short*)alloc(44ull * 320 * 64 * 2);
  unsigned short* agg     = (unsigned short*)alloc(48ull * 320 * 320 * 2);
  unsigned short* g_p     = (unsigned short*)alloc(1232ull * 768 * 2);
  float*          out_het = (float*)alloc(1232ull * 16 * 4);

  hipMemsetAsync(d_ws, 0, off, stream);

  auto cdiv = [](long a, long b) { return (int)((a + b - 1) / b); };

  // weight packing (f32 -> bf16^T, zero padded)
  k_pack_wT<<<cdiv(1824 * 704, 256), 256, 0, stream>>>(W_in, WinT, 1824, 704, 1824, 704);
  k_pack_wT<<<cdiv(64 * 64, 256), 256, 0, stream>>>(Wq, WqT, 64, 64, 64, 64);
  k_pack_wT<<<cdiv(64 * 64, 256), 256, 0, stream>>>(Wk, WkT, 64, 64, 64, 64);
  k_pack_wT<<<cdiv(64 * 64, 256), 256, 0, stream>>>(Wv, WvT, 64, 64, 64, 64);
  k_pack_wT<<<cdiv(64 * 64, 256), 256, 0, stream>>>(Wo, WoT, 64, 64, 64, 64);
  k_pack_wT<<<cdiv(64 * 256, 256), 256, 0, stream>>>(Wf1, Wf1T, 64, 256, 64, 256);
  k_pack_wT<<<cdiv(256 * 64, 256), 256, 0, stream>>>(Wf2, Wf2T, 256, 64, 256, 64);
  k_pack_wT<<<cdiv(768 * 16, 256), 256, 0, stream>>>(Wgo, WgoT, 768, 12, 768, 16);
  k_pack_wgcat<<<cdiv(64 * 320, 256), 256, 0, stream>>>(Wg1, Wg2, WgcatT);
  k_pack_adj<<<cdiv(320 * 320, 256), 256, 0, stream>>>(adj, adjA);
  k_pack_adj<<<cdiv(320 * 320, 256), 256, 0, stream>>>(adjs, adjS);

  // embedding, spline coefficients
  k_embed<<<cdiv(4 * 12 * 307 * 152, 256), 256, 0, stream>>>(ori, W_val, b_val, tod_e, dow_e,
                                                             node_e, e1, e2);
  k_ce<<<cdiv(4 * 307 * 11 * 64, 256), 256, 0, stream>>>(coeff2, W_ce, b_ce, ce);

  // x_in = embedding @ W_in  ->  z state
  k_xin<<<cdiv(77 * 44, 8), 256, 0, stream>>>(e1, WinT, b_in, zst);

  // HeteroGCN branch
  k_agg<<<cdiv(48 * 20 * 10 * 2, 8), 256, 0, stream>>>(adjA, adjS, e2, agg);
  k_gcnproj<<<cdiv(960 * 4, 8), 256, 0, stream>>>(agg, WgcatT, bg, g_p);
  k_gp<<<cdiv(77, 8), 256, 0, stream>>>(g_p, WgoT, bgo, out_het);

  // Euler CDE scan: 11 steps of fused spatial-attention vector field
  const int ATTN_SMEM = 83968 + 40960;   // scores/FFN region + TDM-staged V tile
  for (int i = 0; i < 11; ++i) {
    k_zbf<<<cdiv(4 * 11 * 307 * 64, 256), 256, 0, stream>>>(zst, zbf);
    k_qkv<<<cdiv(4 * 11 * 20 * 4 * 3, 8), 256, 0, stream>>>(zbf, WqT, WkT, WvT, bq, bk, bv,
                                                            qb, kbuf, vbuf);
    k_attn<<<880, 256, ATTN_SMEM, stream>>>(qb, kbuf, vbuf, zst, ce, WoT, Wf1T, Wf2T, bo,
                                            ln1g, ln1b, bf1, bf2, ln2g, ln2b, W_tp2, b_tp2, i);
  }

  // final layernorm + end conv + weighted combine with GCN branch
  k_final<<<cdiv(4 * 307, 256), 256, 0, stream>>>(zst, out_het, lnfg, lnfb, Wend, bend,
                                                  wacl, whet, (float*)d_out);
}
